// ProbSparseSelfAttention_13477607374942
// MI455X (gfx1250) — matrix operand order repaired and checked by reading, not compiled, not run on hardware
//
#include <hip/hip_runtime.h>
#include <stdint.h>

#define D_MODEL 1024
#define N_HEADS 16
#define D_K     64
#define BATCH   2
#define SEQ     2048
#define M_TOTAL (BATCH * SEQ)   // 4096 rows
#define KPANEL  128             // k-elements staged per LDS buffer (4 x 32-k fragments)

typedef __attribute__((ext_vector_type(16))) __bf16 v16bf;
typedef __attribute__((ext_vector_type(8)))  float  v8f;

typedef __attribute__((address_space(3))) const void* lds_cptr_t;

union Frag16 { v16bf v; uint4 u[2]; };   // 32 bytes: one 16x32 bf16 A or 32x16 bf16 B operand (per-lane slice)

static __device__ __forceinline__ v8f zero8() {
  v8f z = {0.f, 0.f, 0.f, 0.f, 0.f, 0.f, 0.f, 0.f};
  return z;
}

static __device__ __forceinline__ unsigned short f32_to_bf16(float f) {
  union { float f; unsigned int u; } cv; cv.f = f;
  unsigned int u = cv.u;
  u += 0x7fffu + ((u >> 16) & 1u);      // round-to-nearest-even
  return (unsigned short)(u >> 16);
}

// ---------------------------------------------------------------------------
// fp32 -> bf16 bulk conversion (n4 = element count / 4)
// ---------------------------------------------------------------------------
__global__ __launch_bounds__(256) void cvt_f32_to_bf16(const float* __restrict__ src,
                                                       unsigned short* __restrict__ dst,
                                                       int n4) {
  int i = blockIdx.x * blockDim.x + threadIdx.x;
  if (i < n4) {
    float4 f = reinterpret_cast<const float4*>(src)[i];
    ushort4 h;
    h.x = f32_to_bf16(f.x); h.y = f32_to_bf16(f.y);
    h.z = f32_to_bf16(f.z); h.w = f32_to_bf16(f.w);
    reinterpret_cast<ushort4*>(dst)[i] = h;
  }
}

// ---------------------------------------------------------------------------
// GEMM: y[m][n] = sum_k A[m][k] * W[n][k] + bias[n]     (y = x @ W^T + b)
// One 16x16 output tile per wave. The weight (B) stream is double-buffered
// through LDS in 128-k panels via GLOBAL_LOAD_ASYNC_TO_LDS_B128; one
// S_WAIT_ASYNCCNT covers 4 back-to-back v_wmma_f32_16x16x32_bf16.
// MODE 0: store bf16, head layout   out[b][h][l][dk]
// MODE 1: store bf16, transposed V  out[b][h][dk][l]
// MODE 2: store f32,  row major     out[m][n]
// Grid: 2048 blocks x 256 threads (8 waves) = 16384 tiles = 256 Mtiles x 64 Ntiles
// ---------------------------------------------------------------------------
template <int MODE>
__global__ __launch_bounds__(256) void gemm_bf16_wmma(const unsigned short* __restrict__ A,
                                                      const unsigned short* __restrict__ W,
                                                      const float* __restrict__ bias,
                                                      void* __restrict__ outp) {
  // per-wave double-buffered 128-k B panel: 2 x (32 lanes x 128B) = 8KB; block total 64KB
  __shared__ __align__(16) unsigned short bstage[8 * 2 * 2048];

  const int lane = threadIdx.x & 31;
  const int wid  = threadIdx.x >> 5;
  const int l2   = lane & 15;        // row (A) / col (B) within tile
  const int hi   = lane >> 4;        // half-wave selector (contiguous K half: k = 16*hi .. 16*hi+15)

  const int tile  = blockIdx.x * 8 + wid;
  const int tileM = tile >> 6;       // 0..255
  const int tileN = tile & 63;       // 0..63

  // A-frag: lane l2 holds row (tileM*16+l2); v0..7 = K hi*16 .. hi*16+15 (contiguous)
  const unsigned short* pa = A + (size_t)(tileM * 16 + l2) * D_MODEL + hi * 16;
  // B-frag: lane l2 holds col n = tileN*16+l2; v0..7 = K hi*16 .. hi*16+15 (contiguous in W row n)
  const unsigned short* pb = W + (size_t)(tileN * 16 + l2) * D_MODEL + hi * 16;

  unsigned short* wb = bstage + wid * 4096;                  // this wave's staging area (elements)
  const unsigned ldsBase = (unsigned)(size_t)(lds_cptr_t)wb; // LDS byte offset of staging area

  // stage the 128-k panel starting at `k` into buffer `buf`:
  // per lane 4 fragments x 32B; LDS per-lane region is 128B.
  // No "memory" clobber here: volatile asms stay ordered w.r.t. the waits below,
  // and this lets the compiler schedule A-loads / B-reads around the issues.
  auto issuePanel = [&](int k, int buf) {
#pragma unroll
    for (int j = 0; j < 4; ++j) {
      unsigned lds = ldsBase + (unsigned)(buf * 4096 + lane * 128 + j * 32);
      const unsigned short* gp = pb + k + j * 32;
      asm volatile("global_load_async_to_lds_b128 %0, %1, off" :: "v"(lds), "v"(gp));
      asm volatile("global_load_async_to_lds_b128 %0, %1, off offset:16" :: "v"(lds), "v"(gp));
    }
  };

  issuePanel(0, 0);
  int cur = 0;

  v8f c = zero8();
  for (int k = 0; k < D_MODEL; k += KPANEL) {
    if (k + KPANEL < D_MODEL) {
      issuePanel(k + KPANEL, cur ^ 1);
      // <=8 outstanding leaves only the next panel's 8 transfers in flight
      // (ASYNCcnt completes in order) => current panel is resident.
      asm volatile("s_wait_asynccnt 0x8" ::: "memory");
    } else {
      asm volatile("s_wait_asynccnt 0x0" ::: "memory");
    }

    __builtin_prefetch(pa + k + 2 * KPANEL, 0, 1);
#pragma unroll
    for (int j = 0; j < 4; ++j) {
      Frag16 a, b;
      a.u[0] = *reinterpret_cast<const uint4*>(pa + k + j * 32);
      a.u[1] = *reinterpret_cast<const uint4*>(pa + k + j * 32 + 8);
      const unsigned short* rb = wb + cur * 2048 + lane * 64 + j * 16;  // per-lane 32B slice
      b.u[0] = *reinterpret_cast<const uint4*>(rb);
      b.u[1] = *reinterpret_cast<const uint4*>(rb + 8);
      c = __builtin_amdgcn_wmma_f32_16x16x32_bf16(false, a.v, false, b.v, (short)0, c, false, false);
    }
    cur ^= 1;
  }

  const int n  = tileN * 16 + l2;
  const float bn = bias[n];

  if (MODE == 2) {
    float* out = (float*)outp;
#pragma unroll
    for (int r = 0; r < 8; ++r) {
      int m = tileM * 16 + hi * 8 + r;         // C layout: VGPR r -> row r (lanes 0-15) / r+8 (lanes 16-31)
      out[(size_t)m * D_MODEL + n] = c[r] + bn;
    }
  } else if (MODE == 0) {
    unsigned short* out = (unsigned short*)outp;
    const int h = n >> 6, d = n & 63;
#pragma unroll
    for (int r = 0; r < 8; ++r) {
      int m = tileM * 16 + hi * 8 + r;
      int b2 = m >> 11, l = m & 2047;
      out[(((size_t)(b2 * N_HEADS + h) * SEQ + l) * D_K) + d] = f32_to_bf16(c[r] + bn);
    }
  } else {
    // transposed V: fixed (h,d) per lane, rows m consecutive over r -> one 16B store
    unsigned short* out = (unsigned short*)outp;
    const int h = n >> 6, d = n & 63;
    const int m0 = tileM * 16 + hi * 8;
    const int b2 = m0 >> 11, l0 = m0 & 2047;
    unsigned short t[8];
#pragma unroll
    for (int r = 0; r < 8; ++r) t[r] = f32_to_bf16(c[r] + bn);
    uint4 pk;
    pk.x = (unsigned)t[0] | ((unsigned)t[1] << 16);
    pk.y = (unsigned)t[2] | ((unsigned)t[3] << 16);
    pk.z = (unsigned)t[4] | ((unsigned)t[5] << 16);
    pk.w = (unsigned)t[6] | ((unsigned)t[7] << 16);
    unsigned short* dst = out + (((size_t)(b2 * N_HEADS + h) * D_K + d) * SEQ) + l0;
    *reinterpret_cast<uint4*>(dst) = pk;
  }
}

// ---------------------------------------------------------------------------
// Flash attention: one 16-row q-tile per wave, streaming over keys in 32-key
// steps. Online softmax in the exp2 domain (scores pre-scaled by 1/8 * log2e).
// P is transposed C-layout -> A-layout through a per-wave LDS tile.
// Grid: 512 blocks x 256 threads (8 waves) = 4096 q-tiles (= 32 bh * 128 qt)
// ---------------------------------------------------------------------------
__global__ __launch_bounds__(256) void attn_wmma(const unsigned short* __restrict__ Q,
                                                 const unsigned short* __restrict__ K,
                                                 const unsigned short* __restrict__ Vt,
                                                 unsigned short* __restrict__ ctx) {
  // per-wave 16x32 bf16 P tile, row stride 40 elems (80B -> 16B-aligned b128 reads)
  __shared__ __align__(16) unsigned short ptile[8 * 16 * 40];

  const int lane = threadIdx.x & 31;
  const int wid  = threadIdx.x >> 5;
  const int l2   = lane & 15;
  const int hi   = lane >> 4;
  unsigned short* myTile = ptile + wid * 640;

  const int idw = blockIdx.x * 8 + wid;
  const int qt  = idw & 127;
  const int bh  = idw >> 7;            // 0..31

  // Q A-fragments for d 0..31 (a0) and d 32..63 (a1)
  const unsigned short* qp = Q + ((size_t)bh * SEQ + qt * 16 + l2) * D_K;
  Frag16 a0, a1;
  a0.u[0] = *(const uint4*)(qp + hi * 16);
  a0.u[1] = *(const uint4*)(qp + hi * 16 + 8);
  a1.u[0] = *(const uint4*)(qp + 32 + hi * 16);
  a1.u[1] = *(const uint4*)(qp + 32 + hi * 16 + 8);

  const float SCALE = 0.125f * 1.44269504088896340736f;  // 1/sqrt(64) * log2(e)

  float mrow[8], lrow[8];
#pragma unroll
  for (int r = 0; r < 8; ++r) { mrow[r] = -__builtin_inff(); lrow[r] = 0.f; }
  v8f o[4];
#pragma unroll
  for (int j = 0; j < 4; ++j) o[j] = zero8();

  const unsigned short* kbase = K  + (size_t)bh * SEQ * D_K;
  const unsigned short* vbase = Vt + (size_t)bh * D_K * SEQ;

#pragma unroll 2
  for (int kt = 0; kt < SEQ; kt += 32) {
    // --- scores: two 16x16 tiles (keys kt..kt+15, kt+16..kt+31), K-dim = d_k = 64 = 2 WMMAs
    v8f s[2];
#pragma unroll
    for (int t = 0; t < 2; ++t) {
      const unsigned short* kp = kbase + (size_t)(kt + t * 16 + l2) * D_K;
      Frag16 b0, b1;
      b0.u[0] = *(const uint4*)(kp + hi * 16);
      b0.u[1] = *(const uint4*)(kp + hi * 16 + 8);
      b1.u[0] = *(const uint4*)(kp + 32 + hi * 16);
      b1.u[1] = *(const uint4*)(kp + 32 + hi * 16 + 8);
      v8f acc = zero8();
      acc = __builtin_amdgcn_wmma_f32_16x16x32_bf16(false, a0.v, false, b0.v, (short)0, acc, false, false);
      acc = __builtin_amdgcn_wmma_f32_16x16x32_bf16(false, a1.v, false, b1.v, (short)0, acc, false, false);
      s[t] = acc;
    }

    // --- online softmax: each row lives in one VGPR across a 16-lane half-wave
    float alpha[8];
#pragma unroll
    for (int r = 0; r < 8; ++r) {
      float s0 = s[0][r] * SCALE;
      float s1 = s[1][r] * SCALE;
      float mx = fmaxf(s0, s1);
#pragma unroll
      for (int off = 1; off < 16; off <<= 1)
        mx = fmaxf(mx, __shfl_xor(mx, off, 16));
      float mn = fmaxf(mrow[r], mx);
      float al = exp2f(mrow[r] - mn);
      float p0 = exp2f(s0 - mn);
      float p1 = exp2f(s1 - mn);
      float ls = p0 + p1;
#pragma unroll
      for (int off = 1; off < 16; off <<= 1)
        ls += __shfl_xor(ls, off, 16);
      lrow[r] = lrow[r] * al + ls;
      mrow[r] = mn;
      alpha[r] = al;
      // stage P (C layout -> LDS row-major): row = hi*8+r, cols l2 and 16+l2
      int row = hi * 8 + r;
      myTile[row * 40 + l2]      = f32_to_bf16(p0);
      myTile[row * 40 + 16 + l2] = f32_to_bf16(p1);
    }

    // rescale running O by alpha (same row mapping as S)
#pragma unroll
    for (int j = 0; j < 4; ++j)
#pragma unroll
      for (int r = 0; r < 8; ++r)
        o[j][r] *= alpha[r];

    // reload P in A-fragment layout (same-wave LDS ops are in-order)
    Frag16 pf;
    pf.u[0] = *(const uint4*)(myTile + l2 * 40 + hi * 16);
    pf.u[1] = *(const uint4*)(myTile + l2 * 40 + hi * 16 + 8);

    // --- O += P(16x32) @ V(32x64): transposed V gives contiguous B-operand loads
#pragma unroll
    for (int j = 0; j < 4; ++j) {
      const unsigned short* vp = vbase + (size_t)(j * 16 + l2) * SEQ + kt + hi * 16;
      Frag16 bv;
      bv.u[0] = *(const uint4*)(vp);
      bv.u[1] = *(const uint4*)(vp + 8);
      o[j] = __builtin_amdgcn_wmma_f32_16x16x32_bf16(false, pf.v, false, bv.v, (short)0, o[j], false, false);
    }
  }

  // normalize and store ctx[b][l][h*64 + d] (bf16, consumed by output projection GEMM)
  const int b = bh >> 4, h = bh & 15;
#pragma unroll
  for (int r = 0; r < 8; ++r) {
    float inv = 1.0f / lrow[r];
    int q = qt * 16 + hi * 8 + r;
    unsigned short* dst = ctx + ((size_t)(b * SEQ + q) * D_MODEL) + h * D_K + l2;
#pragma unroll
    for (int j = 0; j < 4; ++j)
      dst[j * 16] = f32_to_bf16(o[j][r] * inv);
  }
}

// ---------------------------------------------------------------------------
// Host launcher
// ---------------------------------------------------------------------------
extern "C" void kernel_launch(void* const* d_in, const int* in_sizes, int n_in,
                              void* d_out, int out_size, void* d_ws, size_t ws_size,
                              hipStream_t stream) {
  (void)in_sizes; (void)n_in; (void)out_size; (void)ws_size;

  const float* x  = (const float*)d_in[0];
  const float* Wq = (const float*)d_in[1];
  const float* bq = (const float*)d_in[2];
  const float* Wk = (const float*)d_in[3];
  const float* bk = (const float*)d_in[4];
  const float* Wv = (const float*)d_in[5];
  const float* bv = (const float*)d_in[6];
  const float* Wo = (const float*)d_in[7];
  const float* bo = (const float*)d_in[8];

  char* ws = (char*)d_ws;
  const size_t MB = 1024u * 1024u;
  unsigned short* xb  = (unsigned short*)(ws + 0 * MB);   // 8 MB  x bf16
  unsigned short* Wqb = (unsigned short*)(ws + 8 * MB);   // 2 MB
  unsigned short* Wkb = (unsigned short*)(ws + 10 * MB);  // 2 MB
  unsigned short* Wvb = (unsigned short*)(ws + 12 * MB);  // 2 MB
  unsigned short* Wob = (unsigned short*)(ws + 14 * MB);  // 2 MB
  unsigned short* Qb  = (unsigned short*)(ws + 16 * MB);  // 8 MB  [b][h][l][dk]
  unsigned short* Kb  = (unsigned short*)(ws + 24 * MB);  // 8 MB  [b][h][l][dk]
  unsigned short* Vtb = (unsigned short*)(ws + 32 * MB);  // 8 MB  [b][h][dk][l]
  unsigned short* Cxb = (unsigned short*)(ws + 40 * MB);  // 8 MB  [b][l][D]

  cvt_f32_to_bf16<<<(M_TOTAL * D_MODEL / 4 + 255) / 256, 256, 0, stream>>>(x, xb, M_TOTAL * D_MODEL / 4);
  cvt_f32_to_bf16<<<(D_MODEL * D_MODEL / 4 + 255) / 256, 256, 0, stream>>>(Wq, Wqb, D_MODEL * D_MODEL / 4);
  cvt_f32_to_bf16<<<(D_MODEL * D_MODEL / 4 + 255) / 256, 256, 0, stream>>>(Wk, Wkb, D_MODEL * D_MODEL / 4);
  cvt_f32_to_bf16<<<(D_MODEL * D_MODEL / 4 + 255) / 256, 256, 0, stream>>>(Wv, Wvb, D_MODEL * D_MODEL / 4);
  cvt_f32_to_bf16<<<(D_MODEL * D_MODEL / 4 + 255) / 256, 256, 0, stream>>>(Wo, Wob, D_MODEL * D_MODEL / 4);

  gemm_bf16_wmma<0><<<2048, 256, 0, stream>>>(xb, Wqb, bq, Qb);
  gemm_bf16_wmma<0><<<2048, 256, 0, stream>>>(xb, Wkb, bk, Kb);
  gemm_bf16_wmma<1><<<2048, 256, 0, stream>>>(xb, Wvb, bv, Vtb);

  attn_wmma<<<512, 256, 0, stream>>>(Qb, Kb, Vtb, Cxb);

  gemm_bf16_wmma<2><<<2048, 256, 0, stream>>>(Cxb, Wob, bo, (float*)d_out);
}
